// MoleculeEncoder_GAT_82016695485247
// MI455X (gfx1250) — matrix operand order repaired and checked by reading, not compiled, not run on hardware
//
#include <hip/hip_runtime.h>
#include <hip/hip_bf16.h>
#include <math.h>

// ---------------------------------------------------------------------------
// Types for CDNA5 WMMA (wave32, 16x16x32 bf16 -> f32 accumulate)
// ---------------------------------------------------------------------------
typedef __attribute__((ext_vector_type(8)))  float  v8f;
typedef __attribute__((ext_vector_type(16))) __bf16 v16bf;

__device__ __forceinline__ unsigned short bfbits(float f) {
    unsigned u = __float_as_uint(f);
    u += 0x7FFFu + ((u >> 16) & 1u);          // round-to-nearest-even
    return (unsigned short)(u >> 16);
}

__device__ __forceinline__ float atomicMaxFloat(float* addr, float val) {
    unsigned* a = (unsigned*)addr;
    unsigned old = *a;
    while (true) {
        float f = __uint_as_float(old);
        if (f >= val) break;
        unsigned assumed = old;
        old = atomicCAS(a, assumed, __float_as_uint(val));
        if (old == assumed) break;
    }
    return __uint_as_float(old);
}

// self-loop aware edge fetch: edges [0,E) from edge_index, [E,E+N) are loops
__device__ __forceinline__ void edge_sd(const int* __restrict__ ei, int E,
                                        int e, int& s, int& d) {
    if (e < E) { s = ei[e]; d = ei[(size_t)E + e]; }
    else       { s = d = e - E; }
}

// ---------------------------------------------------------------------------
// Fill kernel (workspace init; d_ws is poisoned 0xAA by harness)
// ---------------------------------------------------------------------------
__global__ void fillf_kernel(float* __restrict__ p, float v, int n) {
    int t = blockIdx.x * blockDim.x + threadIdx.x;
    if (t < n) p[t] = v;
}

// ---------------------------------------------------------------------------
// WMMA fragment index helpers (16-bit A 16x32 / B 32x16, wave32; contiguous k order chosen by this code)
//   A: lane = m%16 + 16*half;  element i -> K = i + half*16
//   B: lane = n%16 + 16*half;  element i -> K = i + half*16
//   C: vgpr r, lane l+16*half -> row = r + half*8, col = l
// LDS tiles are stored in *fragment order*: one contiguous v16bf per lane,
// so the fragment load is a single aligned 32-byte DS read.
// ---------------------------------------------------------------------------
__device__ __forceinline__ int a_k_of(int i, int half) {
    return i + half * 16;
}

// ---------------------------------------------------------------------------
// Layer-1 GEMM: xl1 = x @ Wl1 + bl1, xr1 = x @ Wr1 + br1
//   x: [N,54] f32, W*: [54,32].  Fused cols [54(pad64), 64] bf16.
//   8 waves/block, each owns one 16x16 tile (2 row-tiles x 4 col-tiles).
// ---------------------------------------------------------------------------
__global__ __launch_bounds__(256) void gemm1_kernel(
    const float* __restrict__ x,
    const float* __restrict__ Wl, const float* __restrict__ bl,
    const float* __restrict__ Wr, const float* __restrict__ br,
    float* __restrict__ xl1, float* __restrict__ xr1, int N)
{
    __shared__ v16bf ldsA[2][2][32];   // [rt][kb][lane]  (2048 bf16 = 4 KB)
    __shared__ v16bf ldsB[2][4][32];   // [kb][ct][lane]  (4096 bf16 = 8 KB)
    const int tid = threadIdx.x;
    const int rowbase = blockIdx.x * 32;

    // ---- stage A in fragment order: 8 contiguous slots/thread -> 1 uint4 ----
    {
        int j0 = tid * 8;                       // slot = ((rt*2+kb)*32+lane)*16+i
        int i0 = j0 & 15, lane = (j0 >> 4) & 31, kb = (j0 >> 9) & 1, rt = j0 >> 10;
        int half = lane >> 4, l = lane & 15;
        int gr = rowbase + rt * 16 + l;
        unsigned pk[4];
        #pragma unroll
        for (int p = 0; p < 4; ++p) {
            unsigned short w[2];
            #pragma unroll
            for (int q = 0; q < 2; ++q) {
                int i = i0 + 2 * p + q;
                int k = kb * 32 + a_k_of(i, half);
                float v = (gr < N && k < 54) ? x[(size_t)gr * 54 + k] : 0.f;
                w[q] = bfbits(v);
            }
            pk[p] = (unsigned)w[0] | ((unsigned)w[1] << 16);
        }
        uint4* av = reinterpret_cast<uint4*>(&ldsA[0][0][0]);
        uint4 u; u.x = pk[0]; u.y = pk[1]; u.z = pk[2]; u.w = pk[3];
        av[tid] = u;
    }
    // ---- stage B in fragment order: 16 slots/thread -> 2 uint4 ----
    {
        int j0 = tid * 16;                      // slot = ((kb*4+ct)*32+lane)*16+i
        int lane = (j0 >> 4) & 31, ct = (j0 >> 9) & 3, kb = j0 >> 11;
        int half = lane >> 4, l = lane & 15;
        int n = ct * 16 + l;                    // fused col 0..63
        unsigned pk[8];
        #pragma unroll
        for (int p = 0; p < 8; ++p) {
            unsigned short w[2];
            #pragma unroll
            for (int q = 0; q < 2; ++q) {
                int i = 2 * p + q;
                int k = kb * 32 + i + half * 16;
                float v = 0.f;
                if (k < 54) v = (n < 32) ? Wl[k * 32 + n] : Wr[k * 32 + (n - 32)];
                w[q] = bfbits(v);
            }
            pk[p] = (unsigned)w[0] | ((unsigned)w[1] << 16);
        }
        uint4* bv = reinterpret_cast<uint4*>(&ldsB[0][0][0]);
        uint4 u0; u0.x = pk[0]; u0.y = pk[1]; u0.z = pk[2]; u0.w = pk[3];
        uint4 u1; u1.x = pk[4]; u1.y = pk[5]; u1.z = pk[6]; u1.w = pk[7];
        bv[2 * tid]     = u0;
        bv[2 * tid + 1] = u1;
    }
    __syncthreads();

    const int wave = tid >> 5, lane = tid & 31;
    const int half = lane >> 4, l = lane & 15;
    const int rt = wave >> 2;           // row tile 0..1
    const int ct = wave & 3;            // col tile 0..3

    v8f c = {};
    #pragma unroll
    for (int kb = 0; kb < 2; ++kb) {
        v16bf a = ldsA[rt][kb][lane];
        v16bf b = ldsB[kb][ct][lane];
        c = __builtin_amdgcn_wmma_f32_16x16x32_bf16(false, a, false, b,
                                                    (short)0, c, false, false);
    }

    // ---- store: uniform dst/bias per lane, fast path for full tiles ----
    const int col = ct * 16 + l;        // 0..63
    const float bb = (col < 32) ? bl[col] : br[col - 32];
    float* dst = (col < 32) ? (xl1 + col) : (xr1 + (col - 32));
    const int r0 = rowbase + rt * 16 + half * 8;
    if (r0 + 8 <= N) {
        #pragma unroll
        for (int r = 0; r < 8; ++r) dst[(size_t)(r0 + r) * 32] = c[r] + bb;
    } else {
        #pragma unroll
        for (int r = 0; r < 8; ++r)
            if (r0 + r < N) dst[(size_t)(r0 + r) * 32] = c[r] + bb;
    }
}

// ---------------------------------------------------------------------------
// Layer-2 GEMM: xl2 = h @ Wl2 + bl2, xr2 = h @ Wr2 + br2
//   h: [N,32] f32, W*: [32,256].  Fused cols [32, 512]; grid.y = 128-col chunk;
//   8 waves/block each own one 16x16 tile; K=32 -> 1 WMMA.
// ---------------------------------------------------------------------------
__global__ __launch_bounds__(256) void gemm2_kernel(
    const float* __restrict__ h,
    const float* __restrict__ Wl, const float* __restrict__ bl,
    const float* __restrict__ Wr, const float* __restrict__ br,
    float* __restrict__ xl2, float* __restrict__ xr2, int N)
{
    __shared__ v16bf ldsA[32];          // [lane]          (512 bf16 = 1 KB)
    __shared__ v16bf ldsB[8][32];       // [ct][lane]      (4096 bf16 = 8 KB)
    const int tid = threadIdx.x;
    const int rowbase = blockIdx.x * 16;
    const int chunk = blockIdx.y;       // 0..3 -> global cols chunk*128..+127

    // ---- stage A (64 threads x 8 slots) ----
    if (tid < 64) {
        int j0 = tid * 8;               // slot = lane*16 + i
        int i0 = j0 & 15, lane = j0 >> 4;
        int half = lane >> 4, l = lane & 15;
        int gr = rowbase + l;
        unsigned pk[4];
        #pragma unroll
        for (int p = 0; p < 4; ++p) {
            unsigned short w[2];
            #pragma unroll
            for (int q = 0; q < 2; ++q) {
                int i = i0 + 2 * p + q;
                int k = a_k_of(i, half);               // 0..31
                float v = (gr < N) ? h[(size_t)gr * 32 + k] : 0.f;
                w[q] = bfbits(v);
            }
            pk[p] = (unsigned)w[0] | ((unsigned)w[1] << 16);
        }
        uint4* av = reinterpret_cast<uint4*>(&ldsA[0]);
        uint4 u; u.x = pk[0]; u.y = pk[1]; u.z = pk[2]; u.w = pk[3];
        av[tid] = u;
    }
    // ---- stage B (256 threads x 16 slots -> 2 uint4 each) ----
    {
        int j0 = tid * 16;              // slot = (ct*32+lane)*16 + i
        int lane = (j0 >> 4) & 31, ct = j0 >> 9;
        int half = lane >> 4, l = lane & 15;
        int gc = chunk * 128 + ct * 16 + l;            // 0..511
        const float* W = (gc < 256) ? (Wl + gc) : (Wr + (gc - 256));
        unsigned pk[8];
        #pragma unroll
        for (int p = 0; p < 8; ++p) {
            unsigned short w[2];
            #pragma unroll
            for (int q = 0; q < 2; ++q) {
                int i = 2 * p + q;
                int k = i + half * 16;                 // 0..31
                w[q] = bfbits(W[(size_t)k * 256]);
            }
            pk[p] = (unsigned)w[0] | ((unsigned)w[1] << 16);
        }
        uint4* bv = reinterpret_cast<uint4*>(&ldsB[0][0]);
        uint4 u0; u0.x = pk[0]; u0.y = pk[1]; u0.z = pk[2]; u0.w = pk[3];
        uint4 u1; u1.x = pk[4]; u1.y = pk[5]; u1.z = pk[6]; u1.w = pk[7];
        bv[2 * tid]     = u0;
        bv[2 * tid + 1] = u1;
    }
    __syncthreads();

    const int wave = tid >> 5, lane = tid & 31;
    const int half = lane >> 4, l = lane & 15;
    const int ct = wave;                // col tile 0..7 within the 128 cols

    v16bf a = ldsA[lane];
    v16bf b = ldsB[ct][lane];
    v8f c = {};
    c = __builtin_amdgcn_wmma_f32_16x16x32_bf16(false, a, false, b,
                                                (short)0, c, false, false);

    const int gc = chunk * 128 + ct * 16 + l;          // 0..511
    const float bb = (gc < 256) ? bl[gc] : br[gc - 256];
    float* dst = (gc < 256) ? (xl2 + gc) : (xr2 + (gc - 256));
    const int r0 = rowbase + half * 8;
    if (r0 + 8 <= N) {
        #pragma unroll
        for (int r = 0; r < 8; ++r) dst[(size_t)(r0 + r) * 256] = c[r] + bb;
    } else {
        #pragma unroll
        for (int r = 0; r < 8; ++r)
            if (r0 + r < N) dst[(size_t)(r0 + r) * 256] = c[r] + bb;
    }
}

// ---------------------------------------------------------------------------
// Edge kernels
// ---------------------------------------------------------------------------
// layer1 logits: C=8 per head, thread per (edge, head), float4 loads
__global__ void logits1_kernel(const int* __restrict__ ei, int E, int Etot,
                               const float* __restrict__ xl1,
                               const float* __restrict__ xr1,
                               const float* __restrict__ att,
                               float* __restrict__ wlog,
                               float* __restrict__ segm)
{
    int t = blockIdx.x * blockDim.x + threadIdx.x;
    if (t >= Etot * 4) return;
    int e = t >> 2, hh = t & 3;
    int s, d; edge_sd(ei, E, e, s, d);
    const float4* pl = (const float4*)(xl1 + (size_t)s * 32 + hh * 8);
    const float4* pr = (const float4*)(xr1 + (size_t)d * 32 + hh * 8);
    const float4* pa = (const float4*)(att + hh * 8);
    float acc = 0.f;
    #pragma unroll
    for (int i = 0; i < 2; ++i) {
        float4 l = pl[i], r = pr[i], a = pa[i];
        float v;
        v = l.x + r.x; v = v > 0.f ? v : 0.2f * v; acc += v * a.x;
        v = l.y + r.y; v = v > 0.f ? v : 0.2f * v; acc += v * a.y;
        v = l.z + r.z; v = v > 0.f ? v : 0.2f * v; acc += v * a.z;
        v = l.w + r.w; v = v > 0.f ? v : 0.2f * v; acc += v * a.w;
    }
    wlog[t] = acc;
    atomicMaxFloat(&segm[(size_t)d * 4 + hh], acc);
}

// layer2 logits: C=64 per head, float4-vectorized, thread per (edge, head)
__global__ void logits2_kernel(const int* __restrict__ ei, int E, int Etot,
                               const float* __restrict__ xl2,
                               const float* __restrict__ xr2,
                               const float* __restrict__ att,
                               float* __restrict__ wlog,
                               float* __restrict__ segm)
{
    int t = blockIdx.x * blockDim.x + threadIdx.x;
    if (t >= Etot * 4) return;
    int e = t >> 2, hh = t & 3;
    int s, d; edge_sd(ei, E, e, s, d);
    const float4* pl = (const float4*)(xl2 + (size_t)s * 256 + hh * 64);
    const float4* pr = (const float4*)(xr2 + (size_t)d * 256 + hh * 64);
    const float4* pa = (const float4*)(att + hh * 64);
    float acc = 0.f;
    #pragma unroll 4
    for (int i = 0; i < 16; ++i) {
        float4 l = pl[i], r = pr[i], a = pa[i];
        float v;
        v = l.x + r.x; v = v > 0.f ? v : 0.2f * v; acc += v * a.x;
        v = l.y + r.y; v = v > 0.f ? v : 0.2f * v; acc += v * a.y;
        v = l.z + r.z; v = v > 0.f ? v : 0.2f * v; acc += v * a.z;
        v = l.w + r.w; v = v > 0.f ? v : 0.2f * v; acc += v * a.w;
    }
    wlog[t] = acc;
    atomicMaxFloat(&segm[(size_t)d * 4 + hh], acc);
}

// exp pass (layer agnostic): w = exp(logit - segmax); accumulate segment sum
__global__ void edge_exp_kernel(const int* __restrict__ ei, int E, int Etot,
                                const float* __restrict__ segm,
                                float* __restrict__ wlog,
                                float* __restrict__ segs)
{
    int t = blockIdx.x * blockDim.x + threadIdx.x;
    if (t >= Etot * 4) return;
    int e = t >> 2, hh = t & 3;
    int s, d; edge_sd(ei, E, e, s, d); (void)s;
    float m = segm[(size_t)d * 4 + hh];
    if (!(m > -1e37f)) m = 0.f;           // isfinite guard
    float w = __expf(wlog[t] - m);
    wlog[t] = w;
    atomicAdd(&segs[(size_t)d * 4 + hh], w);
}

// layer1 aggregation: thread per (edge, k), k in [0,32)
__global__ void agg1_kernel(const int* __restrict__ ei, int E, int Etot,
                            const float* __restrict__ xl1,
                            const float* __restrict__ wlog,
                            float* __restrict__ acc1)
{
    int t = blockIdx.x * blockDim.x + threadIdx.x;
    if (t >= Etot * 32) return;
    int e = t >> 5, k = t & 31, hh = k >> 3;
    int s, d; edge_sd(ei, E, e, s, d);
    float w = wlog[(size_t)e * 4 + hh];
    atomicAdd(&acc1[(size_t)d * 32 + k], w * xl1[(size_t)s * 32 + k]);
}

// layer1 finalize: divide by segment sum, add bias, relu -> h1
__global__ void finalize1_kernel(const float* __restrict__ acc1,
                                 const float* __restrict__ segs,
                                 const float* __restrict__ b1,
                                 float* __restrict__ h1, int N)
{
    int t = blockIdx.x * blockDim.x + threadIdx.x;
    if (t >= N * 32) return;
    int n = t >> 5, k = t & 31, hh = k >> 3;
    float v = acc1[t] / (segs[(size_t)n * 4 + hh] + 1e-16f) + b1[k];
    h1[t] = fmaxf(v, 0.f);
}

// layer2 aggregation: one block (256 threads) per edge, one col per thread
__global__ __launch_bounds__(256) void agg2_kernel(
    const int* __restrict__ ei, int E, int Etot,
    const float* __restrict__ xl2, const float* __restrict__ wlog,
    float* __restrict__ acc2)
{
    int e = blockIdx.x;
    if (e >= Etot) return;
    int k = threadIdx.x, hh = k >> 6;
    int s, d; edge_sd(ei, E, e, s, d);
    float w = wlog[(size_t)e * 4 + hh];
    atomicAdd(&acc2[(size_t)d * 256 + k], w * xl2[(size_t)s * 256 + k]);
}

// layer2 finalize + mish + graph-mean-pool accumulate: thread per (node, c)
__global__ void finalize2_pool_kernel(const float* __restrict__ acc2,
                                      const float* __restrict__ segs,
                                      const float* __restrict__ b2,
                                      const int* __restrict__ batch,
                                      float* __restrict__ psum, int N)
{
    int t = blockIdx.x * blockDim.x + threadIdx.x;
    if (t >= N * 64) return;
    int n = t >> 6, c = t & 63;
    float v = 0.f;
    #pragma unroll
    for (int hh = 0; hh < 4; ++hh)
        v += acc2[(size_t)n * 256 + hh * 64 + c] /
             (segs[(size_t)n * 4 + hh] + 1e-16f);
    v = v * 0.25f + b2[c];                         // mean over heads + bias
    float sp = (v > 20.f) ? v : log1pf(__expf(v)); // softplus
    float mish = v * tanhf(sp);
    atomicAdd(&psum[(size_t)batch[n] * 64 + c], mish);
}

__global__ void count_kernel(const int* __restrict__ batch,
                             float* __restrict__ pcnt, int N)
{
    int t = blockIdx.x * blockDim.x + threadIdx.x;
    if (t < N) atomicAdd(&pcnt[batch[t]], 1.f);
}

__global__ void divide_kernel(const float* __restrict__ psum,
                              const float* __restrict__ pcnt,
                              float* __restrict__ out, int G)
{
    int t = blockIdx.x * blockDim.x + threadIdx.x;
    if (t >= G * 64) return;
    out[t] = psum[t] / fmaxf(pcnt[t >> 6], 1.f);
}

// ---------------------------------------------------------------------------
// Host launch
// ---------------------------------------------------------------------------
static inline size_t align_up(size_t x) { return (x + 63) & ~(size_t)63; }

extern "C" void kernel_launch(void* const* d_in, const int* in_sizes, int n_in,
                              void* d_out, int out_size, void* d_ws, size_t ws_size,
                              hipStream_t stream)
{
    (void)n_in; (void)ws_size;
    const float* x    = (const float*)d_in[0];
    const int*   ei   = (const int*)  d_in[1];
    const int*   batch= (const int*)  d_in[2];
    const float* Wl1  = (const float*)d_in[3];
    const float* bl1  = (const float*)d_in[4];
    const float* Wr1  = (const float*)d_in[5];
    const float* br1  = (const float*)d_in[6];
    const float* att1 = (const float*)d_in[7];
    const float* b1   = (const float*)d_in[8];
    const float* Wl2  = (const float*)d_in[9];
    const float* bl2  = (const float*)d_in[10];
    const float* Wr2  = (const float*)d_in[11];
    const float* br2  = (const float*)d_in[12];
    const float* att2 = (const float*)d_in[13];
    const float* b2   = (const float*)d_in[14];
    float* out = (float*)d_out;

    const int N    = in_sizes[0] / 54;
    const int E    = in_sizes[1] / 2;
    const int Etot = E + N;
    const int G    = out_size / 64;

    // workspace carve-out (floats)
    float* base = (float*)d_ws;
    size_t off = 0;
    float* xl1  = base + off; off += align_up((size_t)N * 32);
    float* xr1  = base + off; off += align_up((size_t)N * 32);
    float* h1   = base + off; off += align_up((size_t)N * 32);
    float* acc1 = base + off; off += align_up((size_t)N * 32);
    float* xl2  = base + off; off += align_up((size_t)N * 256);
    float* xr2  = base + off; off += align_up((size_t)N * 256);
    float* acc2 = base + off; off += align_up((size_t)N * 256);
    float* wlog = base + off; off += align_up((size_t)Etot * 4);
    float* segm = base + off; off += align_up((size_t)N * 4);
    float* segs = base + off; off += align_up((size_t)N * 4);
    float* psum = base + off; off += align_up((size_t)G * 64);
    float* pcnt = base + off; off += align_up((size_t)G);

    const int T = 256;
    auto blocks = [](long long n, int t) { return (int)((n + t - 1) / t); };
    const float NEG = -INFINITY;

    // ---------------- layer 1 ----------------
    fillf_kernel<<<blocks((long long)N * 4, T), T, 0, stream>>>(segm, NEG, N * 4);
    fillf_kernel<<<blocks((long long)N * 4, T), T, 0, stream>>>(segs, 0.f, N * 4);
    fillf_kernel<<<blocks((long long)N * 32, T), T, 0, stream>>>(acc1, 0.f, N * 32);

    gemm1_kernel<<<blocks(N, 32), T, 0, stream>>>(x, Wl1, bl1, Wr1, br1,
                                                  xl1, xr1, N);
    logits1_kernel<<<blocks((long long)Etot * 4, T), T, 0, stream>>>(
        ei, E, Etot, xl1, xr1, att1, wlog, segm);
    edge_exp_kernel<<<blocks((long long)Etot * 4, T), T, 0, stream>>>(
        ei, E, Etot, segm, wlog, segs);
    agg1_kernel<<<blocks((long long)Etot * 32, T), T, 0, stream>>>(
        ei, E, Etot, xl1, wlog, acc1);
    finalize1_kernel<<<blocks((long long)N * 32, T), T, 0, stream>>>(
        acc1, segs, b1, h1, N);

    // ---------------- layer 2 ----------------
    fillf_kernel<<<blocks((long long)N * 4, T), T, 0, stream>>>(segm, NEG, N * 4);
    fillf_kernel<<<blocks((long long)N * 4, T), T, 0, stream>>>(segs, 0.f, N * 4);
    fillf_kernel<<<blocks((long long)N * 256, T), T, 0, stream>>>(acc2, 0.f, N * 256);

    gemm2_kernel<<<dim3(blocks(N, 16), 4), T, 0, stream>>>(h1, Wl2, bl2,
                                                           Wr2, br2, xl2, xr2, N);
    logits2_kernel<<<blocks((long long)Etot * 4, T), T, 0, stream>>>(
        ei, E, Etot, xl2, xr2, att2, wlog, segm);
    edge_exp_kernel<<<blocks((long long)Etot * 4, T), T, 0, stream>>>(
        ei, E, Etot, segm, wlog, segs);
    agg2_kernel<<<Etot, T, 0, stream>>>(ei, E, Etot, xl2, wlog, acc2);

    // ---------------- pool ----------------
    fillf_kernel<<<blocks((long long)G * 64, T), T, 0, stream>>>(psum, 0.f, G * 64);
    fillf_kernel<<<blocks((long long)G, T), T, 0, stream>>>(pcnt, 0.f, G);
    finalize2_pool_kernel<<<blocks((long long)N * 64, T), T, 0, stream>>>(
        acc2, segs, b2, batch, psum, N);
    count_kernel<<<blocks((long long)N, T), T, 0, stream>>>(batch, pcnt, N);
    divide_kernel<<<blocks((long long)G * 64, T), T, 0, stream>>>(psum, pcnt, out, G);
}
